// AdditiveAttention_51419348467966
// MI455X (gfx1250) — matrix operand order repaired and checked by reading, not compiled, not run on hardware
//
#include <hip/hip_runtime.h>

typedef __bf16 bf16;
typedef bf16  v4bf  __attribute__((ext_vector_type(4)));
typedef bf16  v16bf __attribute__((ext_vector_type(16)));
typedef float v8f   __attribute__((ext_vector_type(8)));

#define T_DIM 2048
#define N_DIM 64
#define E_DIM 1024
#define A_DIM 512
#define M_DIM (T_DIM * N_DIM)      // 131072 flattened (t,n) rows

#define LDS_STRIDE 1032            // 1024 bf16 + 8 pad (row stride 2064B -> even 2-way bank split)
#define ENC_LDS_ELEMS (8 * 16 * LDS_STRIDE)         // 8 wave tiles
#define B_LDS_ELEMS   (16 * LDS_STRIDE)
#define SCORES_LDS_BYTES ((ENC_LDS_ELEMS + B_LDS_ELEMS) * 2)   // 297,216 B < 320KB

union BFPack { uint4 u[2]; v16bf v; };

// Branch-free tanh. Prefer the native gfx1250 V_TANH_F32 when the toolchain
// exposes it; otherwise a straight-line saturating formulation on v_exp_f32
// (no saveexec divergence like the ocml tanhf path).
static __device__ __forceinline__ float fast_tanh(float x) {
#if __has_builtin(__builtin_amdgcn_tanhf)
    return __builtin_amdgcn_tanhf(x);
#elif __has_builtin(__builtin_amdgcn_tanh_f32)
    return __builtin_amdgcn_tanh_f32(x);
#else
    float ax = __builtin_fabsf(x);
    float e  = __expf(2.0f * ax);            // v_exp_f32; inf for large ax is fine
    float r  = 1.0f - 2.0f / (e + 1.0f);     // 2/inf -> 0  => r -> 1 (saturates)
    return __builtin_copysignf(r, x);
#endif
}

// ---------------------------------------------------------------------------
// K1: convert W_e [A, E] f32 -> bf16 (layout preserved; column of B = row of W_e)
// ---------------------------------------------------------------------------
__global__ void k_cvt_we(const float* __restrict__ We, bf16* __restrict__ out) {
    int i = (blockIdx.x * 256 + threadIdx.x) * 4;      // 512 blocks cover 524288 elems
    float4 f = *(const float4*)(We + i);
    v4bf b = { (bf16)f.x, (bf16)f.y, (bf16)f.z, (bf16)f.w };
    *(v4bf*)(out + i) = b;
}

// ---------------------------------------------------------------------------
// K2: d_proj[n,a] = sum_d dec_h[n,d] * W_d[a,d]   (tiny: 67 MFLOP)
// ---------------------------------------------------------------------------
__global__ void k_dproj(const float* __restrict__ dec_h, const float* __restrict__ Wd,
                        float* __restrict__ dProj) {
    int idx = blockIdx.x * 256 + threadIdx.x;          // 0..32767
    int n = idx >> 9, a = idx & (A_DIM - 1);
    const float4* x = (const float4*)(dec_h + (size_t)n * 1024);
    const float4* w = (const float4*)(Wd + (size_t)a * 1024);
    float acc = 0.f;
    #pragma unroll 4
    for (int i = 0; i < 256; ++i) {
        float4 xa = x[i], wa = w[i];
        acc += xa.x * wa.x + xa.y * wa.y + xa.z * wa.z + xa.w * wa.w;
    }
    dProj[idx] = acc;
}

// ---------------------------------------------------------------------------
// K3: scores[m] = sum_a v[a] * tanh(eproj[m,a] + dproj[m%64,a])
// eproj = enc(bf16) x We^T(bf16) via v_wmma_f32_16x16x32_bf16.
// 256 threads = 8 waves; wave w owns rows [blk*128 + w*16, +16).
// ---------------------------------------------------------------------------
__global__ void __launch_bounds__(256)
k_scores(const float* __restrict__ enc, const bf16* __restrict__ weB,
         const float* __restrict__ dProj, const float* __restrict__ vvec,
         float* __restrict__ scores) {
    extern __shared__ bf16 lds[];
    const int tid  = threadIdx.x;
    const int wave = tid >> 5;
    const int lane = tid & 31;
    const int half = lane >> 4;        // K-split half of the wave
    const int col  = lane & 15;        // A-row (lane<16) / B-col; also epilogue column
    const int mW   = blockIdx.x * 128 + wave * 16;   // wave's first global row
    const int n0   = mW & (N_DIM - 1);

    // --- stage this wave's 16x1024 enc rows into LDS as bf16 (coalesced) ---
    bf16* encT = lds + wave * 16 * LDS_STRIDE;
    for (int r = 0; r < 16; ++r) {
        const float* src = enc + (size_t)(mW + r) * E_DIM;
        bf16* dst = encT + r * LDS_STRIDE;
        #pragma unroll
        for (int c = 0; c < 8; ++c) {
            int idx = (c * 32 + lane) * 4;             // 128 floats per step, coalesced
            float4 f = *(const float4*)(src + idx);
            v4bf b = { (bf16)f.x, (bf16)f.y, (bf16)f.z, (bf16)f.w };
            *(v4bf*)(dst + idx) = b;                   // ds_store_b64
        }
    }

    bf16* bT = lds + ENC_LDS_ELEMS;
    float sAcc[8] = {0.f,0.f,0.f,0.f,0.f,0.f,0.f,0.f};

    const int bcol = tid >> 4;                         // 0..15 : B column (= W_e row)
    const int bseg = (tid & 15) * 64;                  // 64 bf16 per thread

    for (int at = 0; at < A_DIM / 16; ++at) {          // 32 A-tiles
        __syncthreads();                               // previous B-tile fully consumed
        {   // cooperative B-tile stage: 16 cols x 1024 K (bf16), col-contiguous in memory
            const bf16* src = weB + (size_t)(at * 16 + bcol) * E_DIM + bseg;
            bf16* dst = bT + bcol * LDS_STRIDE + bseg;
            #pragma unroll
            for (int q = 0; q < 8; ++q)
                *(uint4*)(dst + q * 8) = *(const uint4*)(src + q * 8);
            // pull next tile toward the WGP cache while WMMAs run
            if (at + 1 < A_DIM / 16)
                __builtin_prefetch(weB + (size_t)((at + 1) * 16 + bcol) * E_DIM + bseg, 0, 0);
        }
        __syncthreads();

        v8f c8 = {};                                   // f32 accumulator tile 16x16
        #pragma unroll 4
        for (int k = 0; k < E_DIM / 32; ++k) {         // 32 K-steps
            // A frag: row = lane&15; lane<16 holds K {0..7,16..23}, lane>=16 holds {8..15,24..31}
            const bf16* ap = encT + col * LDS_STRIDE + k * 32 + half * 8;
            BFPack A; A.u[0] = *(const uint4*)ap; A.u[1] = *(const uint4*)(ap + 16);
            // B frag: col = lane&15; lane<16 holds K {0..7,16..23}, lane>=16 holds {8..15,24..31}
            const bf16* bp = bT + col * LDS_STRIDE + k * 32 + half * 8;
            BFPack B; B.u[0] = *(const uint4*)bp; B.u[1] = *(const uint4*)(bp + 16);
            c8 = __builtin_amdgcn_wmma_f32_16x16x32_bf16(
                     false, A.v, false, B.v, (short)0, c8, false, false);
        }

        // fused epilogue: sAcc[r] += v[a] * tanh(C[r] + dproj)   (branch-free tanh)
        int a = at * 16 + col;
        float va = vvec[a];
        #pragma unroll
        for (int r = 0; r < 8; ++r) {                  // C VGPR r: M = r + 8*half, N = col
            int n = n0 + r + 8 * half;
            float val = c8[r] + dProj[n * A_DIM + a];
            sAcc[r] += va * fast_tanh(val);
        }
    }

    // reduce over the 16 columns held across each half-wave
    #pragma unroll
    for (int r = 0; r < 8; ++r) {
        float s = sAcc[r];
        s += __shfl_xor(s, 1); s += __shfl_xor(s, 2);
        s += __shfl_xor(s, 4); s += __shfl_xor(s, 8);
        sAcc[r] = s;
    }
    if (col == 0) {
        #pragma unroll
        for (int r = 0; r < 8; ++r)
            scores[mW + r + 8 * half] = sAcc[r];
    }
}

// ---------------------------------------------------------------------------
// K4: softmax over T for each n; writes alpha to d_out slot
// ---------------------------------------------------------------------------
__global__ void k_softmax(const float* __restrict__ scores, float* __restrict__ alpha) {
    __shared__ float red[256];
    int n = blockIdx.x, tid = threadIdx.x;
    float m = -3.402823466e+38f;
    for (int t = tid; t < T_DIM; t += 256) m = fmaxf(m, scores[t * N_DIM + n]);
    red[tid] = m; __syncthreads();
    for (int s = 128; s > 0; s >>= 1) { if (tid < s) red[tid] = fmaxf(red[tid], red[tid + s]); __syncthreads(); }
    m = red[0]; __syncthreads();
    float sum = 0.f;
    for (int t = tid; t < T_DIM; t += 256) sum += __expf(scores[t * N_DIM + n] - m);
    red[tid] = sum; __syncthreads();
    for (int s = 128; s > 0; s >>= 1) { if (tid < s) red[tid] += red[tid + s]; __syncthreads(); }
    float inv = 1.f / red[0];
    for (int t = tid; t < T_DIM; t += 256)
        alpha[t * N_DIM + n] = __expf(scores[t * N_DIM + n] - m) * inv;
}

// ---------------------------------------------------------------------------
// K5/K6: ctx[n,e] = sum_t alpha[t,n] * enc[t,n,e]  (second HBM pass, split over 16 t-chunks)
// ---------------------------------------------------------------------------
__global__ void k_ctx_part(const float* __restrict__ enc, const float* __restrict__ alpha,
                           float* __restrict__ parts) {
    int b  = blockIdx.x;                 // 4096 blocks
    int n  = b & 63;
    int eb = (b >> 6) & 3;
    int tc = b >> 8;                     // 0..15
    int e  = eb * 256 + threadIdx.x;
    float acc = 0.f;
    #pragma unroll 4
    for (int tt = 0; tt < T_DIM / 16; ++tt) {
        int t = tc * (T_DIM / 16) + tt;
        acc += alpha[t * N_DIM + n] * enc[((size_t)t * N_DIM + n) * E_DIM + e];
    }
    parts[(size_t)tc * (N_DIM * E_DIM) + n * E_DIM + e] = acc;
}

__global__ void k_ctx_reduce(const float* __restrict__ parts, float* __restrict__ ctx) {
    int i = blockIdx.x * 256 + threadIdx.x;            // 0..65535
    float s = 0.f;
    #pragma unroll
    for (int tc = 0; tc < 16; ++tc) s += parts[(size_t)tc * (N_DIM * E_DIM) + i];
    ctx[i] = s;
}

// ---------------------------------------------------------------------------
extern "C" void kernel_launch(void* const* d_in, const int* in_sizes, int n_in,
                              void* d_out, int out_size, void* d_ws, size_t ws_size,
                              hipStream_t stream) {
    (void)in_sizes; (void)n_in; (void)out_size; (void)ws_size;
    const float* enc  = (const float*)d_in[0];   // [T,N,E]
    const float* dech = (const float*)d_in[1];   // [N,D]
    const float* We   = (const float*)d_in[2];   // [A,E]
    const float* Wd   = (const float*)d_in[3];   // [A,D]
    const float* v    = (const float*)d_in[4];   // [1,A]

    char*  ws     = (char*)d_ws;
    bf16*  weB    = (bf16*)(ws + 0);             // 1,048,576 B
    float* dProj  = (float*)(ws + 1048576);      //   131,072 B
    float* scores = (float*)(ws + 1179648);      //   524,288 B
    float* parts  = (float*)(ws + 1703936);      // 4,194,304 B

    float* ctx   = (float*)d_out;                // [64,1024]
    float* alpha = (float*)d_out + N_DIM * E_DIM;// [2048,64]

    k_cvt_we   <<<512,  256, 0, stream>>>(We, weB);
    k_dproj    <<<128,  256, 0, stream>>>(dech, Wd, dProj);
    k_scores   <<<M_DIM / 128, 256, SCORES_LDS_BYTES, stream>>>(enc, weB, dProj, v, scores);
    k_softmax  <<<N_DIM, 256, 0, stream>>>(scores, alpha);
    k_ctx_part <<<4096, 256, 0, stream>>>(enc, alpha, parts);
    k_ctx_reduce<<<256, 256, 0, stream>>>(parts, ctx);
}
